// SioConvLayer_56736517980931
// MI455X (gfx1250) — matrix operand order repaired and checked by reading, not compiled, not run on hardware
//
#include <hip/hip_runtime.h>
#include <hip/hip_bf16.h>

// ---------------- problem constants ----------------
#define BB    4
#define LL    1024
#define DIMD  1024
#define NH    8
#define DD    64
#define RTOK  (BB * LL)        // 4096 token rows
#define KPD   (DIMD / 2)       // 512 bf16-pairs per 1024-K row
#define SC_C  16               // scan chunks
#define SC_T  (LL / SC_C)      // 64 steps per chunk

typedef __attribute__((ext_vector_type(16))) __bf16 v16bf;
typedef __attribute__((ext_vector_type(8)))  float  v8f;
typedef __attribute__((ext_vector_type(4)))  unsigned v4u;
typedef __attribute__((ext_vector_type(8)))  int    v8i;
typedef __attribute__((ext_vector_type(4)))  int    v4i;

union FragU { v4u q[2]; unsigned u[8]; v16bf v; };

#if defined(__gfx1250__) && __has_builtin(__builtin_amdgcn_tensor_load_to_lds)
#define USE_TDM 1
#else
#define USE_TDM 0
#endif
#if __has_include(<hip/amd_detail/amd_gfx1250_TDM.h>)
#define TDM_6ARG 1
#else
#define TDM_6ARG 0
#endif

__device__ inline v8f wmma_bf16(const FragU& a, const FragU& b, v8f c) {
  return __builtin_amdgcn_wmma_f32_16x16x32_bf16(false, a.v, false, b.v,
                                                 (short)0, c, false, false);
}

__device__ inline v8f zero8() {
  v8f z;
  #pragma unroll
  for (int i = 0; i < 8; ++i) z[i] = 0.0f;
  return z;
}

// ---- f32 -> (hi,lo) bf16 split (bf16x3, ~2^-16 relative error) ----
__device__ inline unsigned short bf16_rtne(float x) {
  unsigned u = __builtin_bit_cast(unsigned, x);
  unsigned r = 0x7FFFu + ((u >> 16) & 1u);
  return (unsigned short)((u + r) >> 16);
}
__device__ inline float bf16_to_f32(unsigned short b) {
  return __builtin_bit_cast(float, ((unsigned)b) << 16);
}
__device__ inline void split_pack(float a0, float a1, unsigned& hi, unsigned& lo) {
  unsigned short h0 = bf16_rtne(a0), h1 = bf16_rtne(a1);
  unsigned short l0 = bf16_rtne(a0 - bf16_to_f32(h0));
  unsigned short l1 = bf16_rtne(a1 - bf16_to_f32(h1));
  hi = (unsigned)h0 | ((unsigned)h1 << 16);
  lo = (unsigned)l0 | ((unsigned)l1 << 16);
}

// =====================================================================
// One-time split kernels (remove per-block re-splitting VALU work)
// =====================================================================
// A-layout: row-major f32 (R,K) -> packed pairs (R, K/2)
__global__ __launch_bounds__(256) void split_a_kernel(
    const float* __restrict__ in, unsigned* __restrict__ hi,
    unsigned* __restrict__ lo, int npairs)
{
  const int idx = blockIdx.x * 256 + threadIdx.x;
  if (idx >= npairs) return;
  const float2 v = *(const float2*)(in + (size_t)idx * 2);
  split_pack(v.x, v.y, hi[idx], lo[idx]);
}

// B-transposed layout: w (K,N) row-major -> Bt (N, K/2), pair over K
__global__ __launch_bounds__(256) void split_bT_kernel(
    const float* __restrict__ w, unsigned* __restrict__ hi,
    unsigned* __restrict__ lo, int N, int KP)
{
  const int idx = blockIdx.x * 256 + threadIdx.x;
  if (idx >= N * KP) return;
  const int p = idx % KP, n = idx / KP;     // writes coalesced over p
  const float a = w[(size_t)(2 * p) * N + n];
  const float b = w[(size_t)(2 * p + 1) * N + n];
  split_pack(a, b, hi[idx], lo[idx]);
}

// complex (NH,64,64) interleaved -> 4 packed arrays (NH,64 n,32 p), Bt[n][p]=M[n][2p..2p+1]
__global__ __launch_bounds__(256) void split_mat_kernel(
    const float* __restrict__ M, unsigned* __restrict__ rH, unsigned* __restrict__ rL,
    unsigned* __restrict__ iH, unsigned* __restrict__ iL)
{
  const int idx = blockIdx.x * 256 + threadIdx.x;   // (h*64+n)*32 + p
  if (idx >= NH * 64 * 32) return;
  const float4 q = *(const float4*)(M + (size_t)idx * 4);
  split_pack(q.x, q.z, rH[idx], rL[idx]);
  split_pack(q.y, q.w, iH[idx], iL[idx]);
}

// =====================================================================
// TDM: load a 64-row x 16-dword tile (row stride KP dwords) into LDS
// padded as [64][20] dwords (pad 4 dwords after every 16).
// =====================================================================
#if USE_TDM
__device__ inline void tdm_tile_16x64(const unsigned* gsrc, int KP, void* lds_dst) {
  const unsigned long long ga = (unsigned long long)(size_t)gsrc;
  v4u g0;
  g0.x = 1u;                                           // count=1, user mode
  g0.y = (unsigned)(size_t)lds_dst;                    // LDS byte offset (addr[31:0])
  g0.z = (unsigned)(ga & 0xFFFFFFFFu);                 // global_addr lo
  g0.w = (unsigned)((ga >> 32) & 0x01FFFFFFu) | (2u << 30);  // addr[56:32] | type=2
  v8i g1;
  g1[0] = (int)((2u << 16) | (1u << 20) | (3u << 22) | (3u << 25));
  //          data_size=4B | pad_en     | intrvl=16dw | pad=4dw ; wg_mask=0 (no cluster)
  g1[1] = (int)(16u << 16);        // tensor_dim0 = 16 dwords
  g1[2] = (int)(64u << 16);        // tensor_dim1 = 64 rows
  g1[3] = (int)(16u << 16);        // tile_dim0 = 16 dwords
  g1[4] = (int)64;                 // tile_dim1 = 64 rows
  g1[5] = (int)KP;                 // tensor_dim0_stride (dwords)
  g1[6] = 0;
  g1[7] = 0;
  v4i z = {0, 0, 0, 0};
#if TDM_6ARG
  v8i z8 = {0, 0, 0, 0, 0, 0, 0, 0};
  __builtin_amdgcn_tensor_load_to_lds(g0, g1, z, z, z8, 0);
#else
  __builtin_amdgcn_tensor_load_to_lds(g0, g1, z, z, 0);
#endif
}
#endif

// =====================================================================
// GEMM: C(MxN) = A*B + bias from pre-split operands.
//   Ahg/Alg: (M, K/2) packed pairs.  Bhg/Blg: (N, K/2) packed pairs.
// Block: 64x64 C tile, 128 threads (4 waves).
// TDM path: double-buffered tensor_load_to_lds overlapped with WMMA.
// =====================================================================
__global__ __launch_bounds__(128) void gemm_bf16x3_kernel(
    const unsigned* __restrict__ Ahg, const unsigned* __restrict__ Alg,
    const unsigned* __restrict__ Bhg, const unsigned* __restrict__ Blg,
    const float* __restrict__ bias, float* __restrict__ C, int N, int K)
{
#if USE_TDM
  __shared__ alignas(16) unsigned tiles[2][4][64][20];
#else
  __shared__ alignas(16) unsigned tiles[1][4][64][20];
#endif
  const int KP = K >> 1;
  const int tid = threadIdx.x, lane = tid & 31, wv = tid >> 5;
  const int ln = lane & 15;
  const int hi4 = (lane >> 4) << 2;
  const int hi8 = (lane >> 4) << 3;
  const int m0 = blockIdx.y * 64, n0 = blockIdx.x * 64;

  const unsigned* gsrc[4] = { Ahg + (size_t)m0 * KP, Alg + (size_t)m0 * KP,
                              Bhg + (size_t)n0 * KP, Blg + (size_t)n0 * KP };

  v8f acc[4];
  #pragma unroll
  for (int t = 0; t < 4; ++t) acc[t] = zero8();

  auto compute_tile = [&](int cur) {
    const unsigned (*Ah)[20] = tiles[cur][0];
    const unsigned (*Al)[20] = tiles[cur][1];
    const unsigned (*Bh)[20] = tiles[cur][2];
    const unsigned (*Bl)[20] = tiles[cur][3];
    FragU ah, al;
    const int m = (wv << 4) + ln;
    ah.q[0] = *(const v4u*)&Ah[m][hi4];       // two ds_load_b128 per frag
    ah.q[1] = *(const v4u*)&Ah[m][8 + hi4];
    al.q[0] = *(const v4u*)&Al[m][hi4];
    al.q[1] = *(const v4u*)&Al[m][8 + hi4];
    #pragma unroll
    for (int nt = 0; nt < 4; ++nt) {
      FragU bh, bl;
      const int n = (nt << 4) + ln;
      bh.q[0] = *(const v4u*)&Bh[n][hi4];
      bh.q[1] = *(const v4u*)&Bh[n][8 + hi4];
      bl.q[0] = *(const v4u*)&Bl[n][hi4];
      bl.q[1] = *(const v4u*)&Bl[n][8 + hi4];
      acc[nt] = wmma_bf16(ah, bh, acc[nt]);   // hi*hi
      acc[nt] = wmma_bf16(ah, bl, acc[nt]);   // hi*lo
      acc[nt] = wmma_bf16(al, bh, acc[nt]);   // lo*hi
    }
  };

#if USE_TDM
  if (wv == 0) {
    #pragma unroll
    for (int a2 = 0; a2 < 4; ++a2) tdm_tile_16x64(gsrc[a2], KP, &tiles[0][a2][0][0]);
    __builtin_amdgcn_s_wait_tensorcnt(0);
  }
  __syncthreads();
  int cur = 0;
  for (int k0p = 0; k0p < KP; k0p += 16) {
    if (wv == 0 && k0p + 16 < KP) {          // prefetch next tile via TDM (overlaps WMMA)
      #pragma unroll
      for (int a2 = 0; a2 < 4; ++a2)
        tdm_tile_16x64(gsrc[a2] + (k0p + 16), KP, &tiles[cur ^ 1][a2][0][0]);
    }
    compute_tile(cur);
    if (wv == 0) __builtin_amdgcn_s_wait_tensorcnt(0);
    __syncthreads();
    cur ^= 1;
  }
#else
  for (int k0p = 0; k0p < KP; k0p += 16) {
    for (int idx = tid; idx < 256; idx += 128) {     // 64 rows x 4 v4u chunks
      const int r = idx >> 2, c = (idx & 3) << 2;
      #pragma unroll
      for (int a2 = 0; a2 < 4; ++a2)
        *(v4u*)&tiles[0][a2][r][c] = *(const v4u*)(gsrc[a2] + (size_t)r * KP + k0p + c);
    }
    if (k0p + 16 < KP) {
      __builtin_prefetch(gsrc[tid & 3] + (size_t)(tid >> 2) * KP + k0p + 16, 0, 0);
      __builtin_prefetch(gsrc[tid & 3] + (size_t)(32 + (tid >> 2)) * KP + k0p + 16, 0, 0);
    }
    __syncthreads();
    compute_tile(0);
    __syncthreads();
  }
#endif

  const int rowb = m0 + (wv << 4) + ((lane >> 4) << 3);
  #pragma unroll
  for (int nt = 0; nt < 4; ++nt) {
    const int col = n0 + (nt << 4) + ln;
    const float bv = bias[col];
    #pragma unroll
    for (int r = 0; r < 8; ++r)
      C[(size_t)(rowb + r) * N + col] = acc[nt][r] + bv;
  }
}

// =====================================================================
// prep_a: a_norm[b,l,h] = a * sqrt(mag2)/(1+mag2), a = x@w_a + b_a
// (= a * rsqrt(mag2) * sigmoid(log(mag2)) telescoped; no log/exp)
// =====================================================================
__global__ __launch_bounds__(128) void prep_a_kernel(
    const float* __restrict__ x, const float* __restrict__ w_a,
    const float* __restrict__ b_a, float* __restrict__ a_norm)
{
  __shared__ float xr[DIMD];
  __shared__ float red[16][8];
  const int row = blockIdx.x;
  const float* xp = x + (size_t)row * DIMD;
  for (int i = threadIdx.x; i < DIMD; i += 128) xr[i] = xp[i];
  __syncthreads();
  const int o = threadIdx.x >> 3;
  const int s = threadIdx.x & 7;
  float sum = 0.0f;
  for (int k = s * 128; k < s * 128 + 128; ++k) sum += xr[k] * w_a[k * 16 + o];
  red[o][s] = sum;
  __syncthreads();
  if (threadIdx.x < NH) {
    const int hh = threadIdx.x;
    float re = b_a[2 * hh], im = b_a[2 * hh + 1];
    for (int q = 0; q < 8; ++q) { re += red[2 * hh][q]; im += red[2 * hh + 1][q]; }
    const float mag2 = re * re + im * im;
    const float sc = sqrtf(mag2) / (1.0f + mag2);
    *(float2*)(a_norm + ((size_t)row * NH + hh) * 2) = make_float2(re * sc, im * sc);
  }
}

// =====================================================================
// Complex batched matvec per head: out[l,n] = sum_k M[n][k]*in[l,k]
// B pre-split (per-head (64,32) packed arrays); A split in-kernel (read once).
// Block: 32 rows x 64 cols, 64 threads (2 waves).
// =====================================================================
__global__ __launch_bounds__(64) void cplx_matvec_kernel(
    const float* __restrict__ in,
    const unsigned* __restrict__ mrH, const unsigned* __restrict__ mrL,
    const unsigned* __restrict__ miH, const unsigned* __restrict__ miL,
    float* __restrict__ out)
{
  __shared__ alignas(16) unsigned ArH[32][36], ArL[32][36], AiH[32][36], AiL[32][36];
  __shared__ alignas(16) unsigned BrH[64][36], BrL[64][36], BiH[64][36], BiL[64][36];

  const int tid = threadIdx.x, lane = tid & 31, wv = tid >> 5;
  const int ln = lane & 15;
  const int hi4 = (lane >> 4) << 2;
  const int hi8 = (lane >> 4) << 3;
  const int h = blockIdx.y, l0 = blockIdx.x * 32;

  const size_t mb = (size_t)h * 64 * 32;
  for (int idx = tid; idx < 512; idx += 64) {          // copy pre-split B (4x 8KB)
    const int n = idx >> 3, c = (idx & 7) << 2;
    *(v4u*)&BrH[n][c] = *(const v4u*)(mrH + mb + n * 32 + c);
    *(v4u*)&BrL[n][c] = *(const v4u*)(mrL + mb + n * 32 + c);
    *(v4u*)&BiH[n][c] = *(const v4u*)(miH + mb + n * 32 + c);
    *(v4u*)&BiL[n][c] = *(const v4u*)(miL + mb + n * 32 + c);
  }
  for (int idx = tid; idx < 32 * 32; idx += 64) {      // split A tile (read once)
    const int r = idx >> 5, p = idx & 31;
    const float4 q = *(const float4*)(in + (((size_t)(l0 + r) * NH + h) * 64 + 2 * p) * 2);
    split_pack(q.x, q.z, ArH[r][p], ArL[r][p]);
    split_pack(q.y, q.w, AiH[r][p], AiL[r][p]);
  }
  __syncthreads();

  v8f accR[4], accI[4];
  #pragma unroll
  for (int t = 0; t < 4; ++t) { accR[t] = zero8(); accI[t] = zero8(); }

  #pragma unroll
  for (int kc = 0; kc < 2; ++kc) {
    const int pb = kc << 4;
    FragU arh, arl, aih, ail;
    const int m = (wv << 4) + ln;
    arh.q[0] = *(const v4u*)&ArH[m][pb + hi4]; arh.q[1] = *(const v4u*)&ArH[m][pb + 8 + hi4];
    arl.q[0] = *(const v4u*)&ArL[m][pb + hi4]; arl.q[1] = *(const v4u*)&ArL[m][pb + 8 + hi4];
    aih.q[0] = *(const v4u*)&AiH[m][pb + hi4]; aih.q[1] = *(const v4u*)&AiH[m][pb + 8 + hi4];
    ail.q[0] = *(const v4u*)&AiL[m][pb + hi4]; ail.q[1] = *(const v4u*)&AiL[m][pb + 8 + hi4];
    #pragma unroll
    for (int nt = 0; nt < 4; ++nt) {
      const int n = (nt << 4) + ln;
      FragU brh, brl, bih, bil, nbh, nbl;
      brh.q[0] = *(const v4u*)&BrH[n][pb + hi4]; brh.q[1] = *(const v4u*)&BrH[n][pb + 8 + hi4];
      brl.q[0] = *(const v4u*)&BrL[n][pb + hi4]; brl.q[1] = *(const v4u*)&BrL[n][pb + 8 + hi4];
      bih.q[0] = *(const v4u*)&BiH[n][pb + hi4]; bih.q[1] = *(const v4u*)&BiH[n][pb + 8 + hi4];
      bil.q[0] = *(const v4u*)&BiL[n][pb + hi4]; bil.q[1] = *(const v4u*)&BiL[n][pb + 8 + hi4];
      #pragma unroll
      for (int j = 0; j < 8; ++j) {
        nbh.u[j] = bih.u[j] ^ 0x80008000u;    // -Bi (f32-dest WMMA has no A/B NEG)
        nbl.u[j] = bil.u[j] ^ 0x80008000u;
      }
      v8f aR = accR[nt];                       // Re = Ar*Br - Ai*Bi
      aR = wmma_bf16(arh, brh, aR);
      aR = wmma_bf16(arh, brl, aR);
      aR = wmma_bf16(arl, brh, aR);
      aR = wmma_bf16(aih, nbh, aR);
      aR = wmma_bf16(aih, nbl, aR);
      aR = wmma_bf16(ail, nbh, aR);
      accR[nt] = aR;
      v8f aI = accI[nt];                       // Im = Ar*Bi + Ai*Br
      aI = wmma_bf16(arh, bih, aI);
      aI = wmma_bf16(arh, bil, aI);
      aI = wmma_bf16(arl, bih, aI);
      aI = wmma_bf16(aih, brh, aI);
      aI = wmma_bf16(aih, brl, aI);
      aI = wmma_bf16(ail, brh, aI);
      accI[nt] = aI;
    }
  }

  const int rowb = l0 + (wv << 4) + ((lane >> 4) << 3);
  #pragma unroll
  for (int nt = 0; nt < 4; ++nt) {
    const int n = (nt << 4) + ln;
    #pragma unroll
    for (int r = 0; r < 8; ++r)
      *(float2*)(out + (((size_t)(rowb + r) * NH + h) * 64 + n) * 2) =
          make_float2(accR[nt][r], accI[nt][r]);
  }
}

// =====================================================================
// Chunked parallel scan of h[l] = vx[l] + a[l]*h[l-1]  (complex, per b,h,i)
// =====================================================================
__global__ __launch_bounds__(64) void scan_phase1_kernel(
    const float* __restrict__ vx, const float* __restrict__ a_norm,
    float* __restrict__ hbuf, float* __restrict__ cp)
{
  const int c = blockIdx.x, bh = blockIdx.y;
  const int b = bh >> 3, h = bh & 7;
  const int i = threadIdx.x;
  float hr = 0.0f, hi = 0.0f, pr = 1.0f, pi = 0.0f;
  for (int t = 0; t < SC_T; ++t) {
    const int l = c * SC_T + t;
    const size_t ridx = (size_t)(b * LL + l) * NH + h;
    const float2 av = *(const float2*)(a_norm + ridx * 2);
    const float2 vv = *(const float2*)(vx + (ridx * 64 + i) * 2);
    const float nr = vv.x + av.x * hr - av.y * hi;
    const float ni = vv.y + av.x * hi + av.y * hr;
    hr = nr; hi = ni;
    *(float2*)(hbuf + (ridx * 64 + i) * 2) = make_float2(hr, hi);
    if (i == 0) {
      const float qr = pr * av.x - pi * av.y;
      const float qi = pr * av.y + pi * av.x;
      pr = qr; pi = qi;
      *(float2*)(cp + ((size_t)bh * LL + l) * 2) = make_float2(pr, pi);
    }
  }
}

__global__ __launch_bounds__(64) void scan_phase2_kernel(
    const float* __restrict__ hidden, const float* __restrict__ hbuf,
    const float* __restrict__ cp, float* __restrict__ carry,
    float* __restrict__ hidden_next)
{
  const int bh = blockIdx.x;
  const int b = bh >> 3, h = bh & 7;
  const int i = threadIdx.x;
  const float2 cv = *(const float2*)(hidden + ((size_t)bh * 64 + i) * 2);
  float cr = cv.x, ci = cv.y;
  for (int c = 0; c < SC_C; ++c) {
    *(float2*)(carry + (((size_t)bh * SC_C + c) * 64 + i) * 2) = make_float2(cr, ci);
    const int lend = c * SC_T + SC_T - 1;
    const float2 Ac = *(const float2*)(cp + ((size_t)bh * LL + lend) * 2);
    const float2 le = *(const float2*)(hbuf + (((size_t)(b * LL + lend) * NH + h) * 64 + i) * 2);
    const float nr = le.x + Ac.x * cr - Ac.y * ci;
    const float ni = le.y + Ac.x * ci + Ac.y * cr;
    cr = nr; ci = ni;
  }
  *(float2*)(hidden_next + ((size_t)bh * 64 + i) * 2) = make_float2(cr, ci);
}

__global__ __launch_bounds__(256) void scan_phase3_kernel(
    float* __restrict__ hbuf, const float* __restrict__ cp,
    const float* __restrict__ carry)
{
  const size_t idx = (size_t)blockIdx.x * 256 + threadIdx.x;
  const size_t i = idx & 63;
  const size_t rh = idx >> 6;
  const size_t h = rh & 7;
  const size_t bl = rh >> 3;
  const size_t l = bl & (LL - 1);
  const size_t b = bl >> 10;
  const size_t c = l / SC_T;
  const size_t bh = b * NH + h;
  const float2 loc = *(const float2*)(hbuf + idx * 2);
  const float2 p = *(const float2*)(cp + (bh * LL + l) * 2);
  const float2 cv = *(const float2*)(carry + ((bh * SC_C + c) * 64 + i) * 2);
  const float nr = loc.x + p.x * cv.x - p.y * cv.y;
  const float ni = loc.y + p.x * cv.y + p.y * cv.x;
  *(float2*)(hbuf + idx * 2) = make_float2(nr, ni);
}

// =====================================================================
// LayerNorm over 128 per (b,l,h); emits PACKED bf16 hi/lo pairs for GEMM2.
// =====================================================================
__global__ __launch_bounds__(128) void ln_kernel(
    const float* __restrict__ h2, const float* __restrict__ gamma,
    const float* __restrict__ beta, unsigned* __restrict__ hnH,
    unsigned* __restrict__ hnL)
{
  __shared__ float ssum[128], ssq[128], vals[128];
  const int g = blockIdx.x;            // (b*LL+l)*NH + h
  const int j = threadIdx.x;
  const float v = h2[(size_t)g * 128 + j];
  ssum[j] = v; ssq[j] = v * v;
  __syncthreads();
  for (int st = 64; st > 0; st >>= 1) {
    if (j < st) { ssum[j] += ssum[j + st]; ssq[j] += ssq[j + st]; }
    __syncthreads();
  }
  const float mu = ssum[0] * (1.0f / 128.0f);
  const float var = ssq[0] * (1.0f / 128.0f) - mu * mu;
  const float rs = rsqrtf(var + 1e-5f);
  vals[j] = (v - mu) * rs * gamma[j] + beta[j];
  __syncthreads();
  if (j < 64) {
    unsigned hi, lo;
    split_pack(vals[2 * j], vals[2 * j + 1], hi, lo);
    const size_t row = (size_t)g >> 3;
    const int hh = g & 7;
    hnH[row * KPD + hh * 64 + j] = hi;
    hnL[row * KPD + hh * 64 + j] = lo;
  }
}

// =====================================================================
extern "C" void kernel_launch(void* const* d_in, const int* in_sizes, int n_in,
                              void* d_out, int out_size, void* d_ws, size_t ws_size,
                              hipStream_t stream) {
  (void)in_sizes; (void)n_in; (void)out_size; (void)ws_size;
  const float* x      = (const float*)d_in[0];
  const float* hidden = (const float*)d_in[1];
  const float* w_x    = (const float*)d_in[2];
  const float* b_x    = (const float*)d_in[3];
  const float* w_a    = (const float*)d_in[4];
  const float* b_a    = (const float*)d_in[5];
  const float* w_y    = (const float*)d_in[6];
  const float* b_y    = (const float*)d_in[7];
  const float* gamma  = (const float*)d_in[8];
  const float* beta   = (const float*)d_in[9];
  const float* mat_v  = (const float*)d_in[10];
  const float* mat_w  = (const float*)d_in[11];

  float* out_y      = (float*)d_out;
  float* out_hidden = out_y + (size_t)RTOK * DIMD;

  // workspace layout (4-byte words)
  float* ws   = (float*)d_ws;
  float* xp   = ws;                                  // 4096*1024 f32
  float* vx   = xp + (size_t)RTOK * DIMD;
  float* hbuf = vx + (size_t)RTOK * DIMD;
  unsigned* xsH = (unsigned*)(hbuf + (size_t)RTOK * DIMD);   // (4096,512) pairs
  unsigned* xsL = xsH + (size_t)RTOK * KPD;
  unsigned* wTH = xsL + (size_t)RTOK * KPD;          // (1024,512) pairs (shared w_x/w_y)
  unsigned* wTL = wTH + (size_t)DIMD * KPD;
  unsigned* mwS = wTL + (size_t)DIMD * KPD;          // mat_w: 4 x (8,64,32)
  unsigned* mvS = mwS + (size_t)4 * NH * 64 * 32;    // mat_v: 4 x (8,64,32)
  float* an    = (float*)(mvS + (size_t)4 * NH * 64 * 32);
  float* cp    = an + (size_t)RTOK * NH * 2;
  float* carry = cp + (size_t)BB * NH * LL * 2;
  const size_t MSZ = (size_t)NH * 64 * 32;
  unsigned* hnH = xsH;   // x-split dead after GEMM1
  unsigned* hnL = xsL;
  float* h2 = xp;        // xp dead after first matvec

  // --- one-time splits ---
  split_a_kernel<<<(RTOK * KPD) / 256, 256, 0, stream>>>(x, xsH, xsL, RTOK * KPD);
  split_bT_kernel<<<(DIMD * KPD) / 256, 256, 0, stream>>>(w_x, wTH, wTL, DIMD, KPD);
  split_mat_kernel<<<(NH * 64 * 32) / 256, 256, 0, stream>>>(
      mat_w, mwS, mwS + MSZ, mwS + 2 * MSZ, mwS + 3 * MSZ);
  split_mat_kernel<<<(NH * 64 * 32) / 256, 256, 0, stream>>>(
      mat_v, mvS, mvS + MSZ, mvS + 2 * MSZ, mvS + 3 * MSZ);

  // 1) xp = x @ w_x + b_x
  gemm_bf16x3_kernel<<<dim3(DIMD / 64, RTOK / 64), 128, 0, stream>>>(
      xsH, xsL, wTH, wTL, b_x, xp, DIMD, DIMD);
  // 2) a_norm
  prep_a_kernel<<<RTOK, 128, 0, stream>>>(x, w_a, b_a, an);
  // 3) vx = mat_w matvec
  cplx_matvec_kernel<<<dim3(RTOK / 32, NH), 64, 0, stream>>>(
      xp, mwS, mwS + MSZ, mwS + 2 * MSZ, mwS + 3 * MSZ, vx);
  // 4-6) chunked scan (+ hidden_next)
  scan_phase1_kernel<<<dim3(SC_C, BB * NH), 64, 0, stream>>>(vx, an, hbuf, cp);
  scan_phase2_kernel<<<BB * NH, 64, 0, stream>>>(hidden, hbuf, cp, carry, out_hidden);
  scan_phase3_kernel<<<(RTOK * NH * DD) / 256, 256, 0, stream>>>(hbuf, cp, carry);
  // 7) h2 = mat_v matvec
  cplx_matvec_kernel<<<dim3(RTOK / 32, NH), 64, 0, stream>>>(
      hbuf, mvS, mvS + MSZ, mvS + 2 * MSZ, mvS + 3 * MSZ, h2);
  // 8) LayerNorm -> packed hn
  ln_kernel<<<RTOK * NH, 128, 0, stream>>>(h2, gamma, beta, hnH, hnL);
  // 9) y = hn @ w_y + b_y
  split_bT_kernel<<<(DIMD * KPD) / 256, 256, 0, stream>>>(w_y, wTH, wTL, DIMD, KPD);
  gemm_bf16x3_kernel<<<dim3(DIMD / 64, RTOK / 64), 128, 0, stream>>>(
      hnH, hnL, wTH, wTL, b_y, out_y, DIMD, DIMD);
}
